// SSMyz_2104533975586
// MI455X (gfx1250) — matrix operand order repaired and checked by reading, not compiled, not run on hardware
//
#include <hip/hip_runtime.h>

// ---------------- types ----------------
typedef _Float16 f16;
typedef _Float16 v16h __attribute__((ext_vector_type(16)));
typedef _Float16 v8h  __attribute__((ext_vector_type(8)));
typedef float    v8f  __attribute__((ext_vector_type(8)));
typedef float    v4f  __attribute__((ext_vector_type(4)));

// ---------------- problem constants ----------------
#define BATCH   2
#define LEN     2048
#define DMODEL  1024
#define DINNER  2048
#define NTOK    (BATCH * LEN)       // 4096 tokens
#define NXZ     (2 * DINNER)        // 4096 outputs of in_proj
#define DTRANK  64
#define DSTATE  16
#define KDBL    (DTRANK + 2 * DSTATE)   // 96

// ---------------- WMMA helpers (contiguous k order for A and B, wave32) ----------------
// A (16xK f16): lanes 0-15 hold row M=lane, K = k0..k0+15;
// lanes 16-31 hold the same rows, K = k0+16..k0+31. Two contiguous v8h runs.
__device__ __forceinline__ v16h wmma_load_a(const f16* __restrict__ A,
                                            int stride, int t0, int k0, int lane) {
    const int lm = lane & 15;
    const int hs = (lane >> 4) << 4;          // 0 or 16
    const f16* p = A + (size_t)(t0 + lm) * stride + k0 + hs;
    v8h lo = *(const v8h*)p;                  // K = k0+hs .. k0+hs+7
    v8h hi = *(const v8h*)(p + 8);            // K = k0+hs+8 .. k0+hs+15
    v16h a;
#pragma unroll
    for (int i = 0; i < 8; ++i) { a[i] = lo[i]; a[8 + i] = hi[i]; }
    return a;
}

// B (Kx16 f16, column n comes from weight row n): lanes 0-15 hold col N=lane,
// K = k0..k0+15; lanes 16-31 hold K = k0+16..k0+31. One contiguous v16h run.
__device__ __forceinline__ v16h wmma_load_b(const f16* __restrict__ W,
                                            int stride, int n0, int k0, int lane) {
    const int lm = lane & 15;
    const int ko = (lane >> 4) << 4;          // 0 or 16
    return *(const v16h*)(W + (size_t)(n0 + lm) * stride + k0 + ko);
}

#define WMMA_F16(a, b, c) \
    __builtin_amdgcn_wmma_f32_16x16x32_f16(false, (a), false, (b), (short)0, (c), false, false)

// ---------------- kernel 0: f32 -> f16 convert ----------------
__global__ void cvt_f32_f16(const float* __restrict__ s, f16* __restrict__ d, int n) {
    int i = blockIdx.x * blockDim.x + threadIdx.x;
    if (i < n) d[i] = (f16)s[i];
}

// ---------------- kernel 1: in_proj GEMM (2-deep pipeline, spill-free) ----------------
// C[t][e] = sum_d hs[t][d] * W[e][d];  e<DINNER -> x (b,d,l), else -> z (b,d,l) + zT (t,d)
__global__ __launch_bounds__(256)
__attribute__((amdgpu_waves_per_eu(2)))
void gemm_inproj(const f16* __restrict__ A16, const f16* __restrict__ W16,
                 float* __restrict__ xbuf, float* __restrict__ zout,
                 float* __restrict__ zT) {
    const int wid  = (blockIdx.x * blockDim.x + threadIdx.x) >> 5;
    const int lane = threadIdx.x & 31;
    const int mT = wid >> 5;                  // 256 token tiles
    const int nT = wid & 31;                  // 32 tiles of 128 outputs
    const int t0 = mT * 16;
    const int n0 = nT * 128;

    v8f acc[8] = {};
    v16h aBuf[2];
    v16h bBuf[2][8];

    // prologue: stage k=0 tiles into slot 0
    aBuf[0] = wmma_load_a(A16, DMODEL, t0, 0, lane);
#pragma unroll
    for (int j = 0; j < 8; ++j)
        bBuf[0][j] = wmma_load_b(W16, DMODEL, n0 + j * 16, 0, lane);

    // steady state: branch-free, symmetric 2-deep pipeline, two k-steps per iter
#pragma unroll 1
    for (int k0 = 0; k0 < DMODEL - 64; k0 += 64) {
        aBuf[1] = wmma_load_a(A16, DMODEL, t0, k0 + 32, lane);
#pragma unroll
        for (int j = 0; j < 8; ++j)
            bBuf[1][j] = wmma_load_b(W16, DMODEL, n0 + j * 16, k0 + 32, lane);
#pragma unroll
        for (int j = 0; j < 8; ++j)
            acc[j] = WMMA_F16(aBuf[0], bBuf[0][j], acc[j]);
        aBuf[0] = wmma_load_a(A16, DMODEL, t0, k0 + 64, lane);
#pragma unroll
        for (int j = 0; j < 8; ++j)
            bBuf[0][j] = wmma_load_b(W16, DMODEL, n0 + j * 16, k0 + 64, lane);
#pragma unroll
        for (int j = 0; j < 8; ++j)
            acc[j] = WMMA_F16(aBuf[1], bBuf[1][j], acc[j]);
    }
    // epilogue: last two k-steps (k = DMODEL-64, DMODEL-32)
    {
        aBuf[1] = wmma_load_a(A16, DMODEL, t0, DMODEL - 32, lane);
#pragma unroll
        for (int j = 0; j < 8; ++j)
            bBuf[1][j] = wmma_load_b(W16, DMODEL, n0 + j * 16, DMODEL - 32, lane);
#pragma unroll
        for (int j = 0; j < 8; ++j)
            acc[j] = WMMA_F16(aBuf[0], bBuf[0][j], acc[j]);
#pragma unroll
        for (int j = 0; j < 8; ++j)
            acc[j] = WMMA_F16(aBuf[1], bBuf[1][j], acc[j]);
    }

    // C/D layout: lane<16 -> N=lane, M=r ; lane>=16 -> N=lane-16, M=r+8
    const int lm  = lane & 15;
    const int hs  = (lane >> 4) << 3;         // 0 or 8
    const int bI  = t0 >> 11;                 // batch
    const int l0  = (t0 & (LEN - 1)) + hs;    // seq position of acc element 0
#pragma unroll
    for (int j = 0; j < 8; ++j) {
        const int e = n0 + j * 16 + lm;
        v4f lo = __builtin_shufflevector(acc[j], acc[j], 0, 1, 2, 3);
        v4f hi = __builtin_shufflevector(acc[j], acc[j], 4, 5, 6, 7);
        if (e < DINNER) {
            float* dst = xbuf + ((size_t)(bI * DINNER + e)) * LEN + l0;
            *(v4f*)dst = lo; *(v4f*)(dst + 4) = hi;
        } else {
            const int ez = e - DINNER;
            float* dst = zout + ((size_t)(bI * DINNER + ez)) * LEN + l0;
            *(v4f*)dst = lo; *(v4f*)(dst + 4) = hi;
#pragma unroll
            for (int r = 0; r < 8; ++r)       // transposed copy for scan
                zT[(size_t)(t0 + hs + r) * DINNER + ez] = acc[j][r];
        }
    }
}

// ---------------- kernel 2: causal depthwise conv + SiLU (LDS transpose) ----------------
__global__ void conv_silu(const float* __restrict__ x, const float* __restrict__ cw,
                          const float* __restrict__ cb,
                          float* __restrict__ xc32, f16* __restrict__ xc16) {
    __shared__ float tile[32][36];
    const int l0 = blockIdx.x * 32;
    const int d0 = blockIdx.y * 32;
    const int b  = blockIdx.z;

    for (int i = threadIdx.x; i < 32 * 35; i += 256) {
        const int di = i / 35, li = i % 35;
        const int gl = l0 - 3 + li;
        float v = 0.f;
        if (gl >= 0) v = x[((size_t)(b * DINNER + d0 + di)) * LEN + gl];
        tile[di][li] = v;
    }
    __syncthreads();

    for (int i = threadIdx.x; i < 1024; i += 256) {
        const int di = i & 31, li = i >> 5;
        const int d = d0 + di;
        float s = cb[d];
#pragma unroll
        for (int w = 0; w < 4; ++w) s += cw[d * 4 + w] * tile[di][li + w];
        const float sig = 1.f / (1.f + __expf(-s));
        const float v = s * sig;
        const size_t o = ((size_t)(b * LEN + l0 + li)) * DINNER + d;   // (token, d)
        xc32[o] = v;
        xc16[o] = (f16)v;
    }
}

// ---------------- kernel 3: x_proj GEMM (4096 x 96, K=2048, pipelined) ----------------
__global__ __launch_bounds__(256)
__attribute__((amdgpu_waves_per_eu(2)))
void gemm_xproj(const f16* __restrict__ A16, const f16* __restrict__ W16,
                float* __restrict__ xdbl32, f16* __restrict__ xdbl16) {
    const int wid  = (blockIdx.x * blockDim.x + threadIdx.x) >> 5;   // 256 waves
    const int lane = threadIdx.x & 31;
    const int t0 = wid * 16;

    v8f acc[6] = {};
    v16h aBuf[2];
    v16h bBuf[2][6];

    aBuf[0] = wmma_load_a(A16, DINNER, t0, 0, lane);
#pragma unroll
    for (int j = 0; j < 6; ++j)
        bBuf[0][j] = wmma_load_b(W16, DINNER, j * 16, 0, lane);

#pragma unroll 1
    for (int k0 = 0; k0 < DINNER - 64; k0 += 64) {
        aBuf[1] = wmma_load_a(A16, DINNER, t0, k0 + 32, lane);
#pragma unroll
        for (int j = 0; j < 6; ++j)
            bBuf[1][j] = wmma_load_b(W16, DINNER, j * 16, k0 + 32, lane);
#pragma unroll
        for (int j = 0; j < 6; ++j)
            acc[j] = WMMA_F16(aBuf[0], bBuf[0][j], acc[j]);
        aBuf[0] = wmma_load_a(A16, DINNER, t0, k0 + 64, lane);
#pragma unroll
        for (int j = 0; j < 6; ++j)
            bBuf[0][j] = wmma_load_b(W16, DINNER, j * 16, k0 + 64, lane);
#pragma unroll
        for (int j = 0; j < 6; ++j)
            acc[j] = WMMA_F16(aBuf[1], bBuf[1][j], acc[j]);
    }
    {
        aBuf[1] = wmma_load_a(A16, DINNER, t0, DINNER - 32, lane);
#pragma unroll
        for (int j = 0; j < 6; ++j)
            bBuf[1][j] = wmma_load_b(W16, DINNER, j * 16, DINNER - 32, lane);
#pragma unroll
        for (int j = 0; j < 6; ++j)
            acc[j] = WMMA_F16(aBuf[0], bBuf[0][j], acc[j]);
#pragma unroll
        for (int j = 0; j < 6; ++j)
            acc[j] = WMMA_F16(aBuf[1], bBuf[1][j], acc[j]);
    }

    const int lm = lane & 15;
    const int hs = (lane >> 4) << 3;
#pragma unroll
    for (int j = 0; j < 6; ++j) {
        const int n = j * 16 + lm;
#pragma unroll
        for (int r = 0; r < 8; ++r) {
            const int tok = t0 + hs + r;
            const float v = acc[j][r];
            xdbl32[(size_t)tok * KDBL + n] = v;
            if (n < DTRANK) xdbl16[(size_t)tok * DTRANK + n] = (f16)v;
        }
    }
}

// ---------------- kernel 4: dt_proj GEMM (4096 x 2048, K=64) ----------------
__global__ __launch_bounds__(256)
__attribute__((amdgpu_waves_per_eu(2)))
void gemm_dt(const f16* __restrict__ A16, const f16* __restrict__ W16,
             float* __restrict__ delta_pre) {
    const int wid  = (blockIdx.x * blockDim.x + threadIdx.x) >> 5;   // 4096 waves
    const int lane = threadIdx.x & 31;
    const int mT = wid >> 4;
    const int nT = wid & 15;
    const int t0 = mT * 16;
    const int n0 = nT * 128;

    v8f acc[8] = {};
#pragma unroll
    for (int k0 = 0; k0 < DTRANK; k0 += 32) {
        v16h a = wmma_load_a(A16, DTRANK, t0, k0, lane);
#pragma unroll
        for (int j = 0; j < 8; ++j) {
            v16h b = wmma_load_b(W16, DTRANK, n0 + j * 16, k0, lane);
            acc[j] = WMMA_F16(a, b, acc[j]);
        }
    }

    const int lm = lane & 15;
    const int hs = (lane >> 4) << 3;
#pragma unroll
    for (int j = 0; j < 8; ++j) {
        const int n = n0 + j * 16 + lm;
#pragma unroll
        for (int r = 0; r < 8; ++r)
            delta_pre[(size_t)(t0 + hs + r) * DINNER + n] = acc[j][r];
    }
}

// ---------------- kernel 5: selective scan + gating ----------------
__global__ void scan_kernel(const float* __restrict__ delta_pre, const float* __restrict__ u32,
                            const float* __restrict__ xdbl32, const float* __restrict__ zT,
                            const float* __restrict__ dtb, const float* __restrict__ A_log,
                            const float* __restrict__ Dp, float* __restrict__ out) {
    const int tid = blockIdx.x * blockDim.x + threadIdx.x;
    if (tid >= BATCH * DINNER) return;
    const int b = tid >> 11;
    const int d = tid & (DINNER - 1);

    float Arow[DSTATE];
#pragma unroll
    for (int n = 0; n < DSTATE; ++n) Arow[n] = -__expf(A_log[d * DSTATE + n]);
    const float dtbias = dtb[d];
    const float Dv = Dp[d];

    float h[DSTATE];
#pragma unroll
    for (int n = 0; n < DSTATE; ++n) h[n] = 0.f;

    for (int l = 0; l < LEN; ++l) {
        const size_t t = (size_t)b * LEN + l;
        const float xdt = delta_pre[t * DINNER + d] + dtbias;
        const float delta = (xdt > 20.f) ? xdt : log1pf(__expf(xdt));   // softplus
        const float u  = u32[t * DINNER + d];
        const float du = delta * u;
        const float* bc = xdbl32 + t * KDBL;
        float y = 0.f;
#pragma unroll
        for (int n = 0; n < DSTATE; ++n) {
            const float dA = __expf(delta * Arow[n]);
            h[n] = h[n] * dA + du * bc[DTRANK + n];          // B_t
            y += h[n] * bc[DTRANK + DSTATE + n];             // C_t
        }
        y += Dv * u;
        const float zv = zT[t * DINNER + d];
        const float yz = y * zv * (1.f / (1.f + __expf(-zv)));  // y * silu(z)
        out[t * DINNER + d] = yz;                            // (b, l, d_inner)
    }
}

// ---------------- launch ----------------
extern "C" void kernel_launch(void* const* d_in, const int* in_sizes, int n_in,
                              void* d_out, int out_size, void* d_ws, size_t ws_size,
                              hipStream_t stream) {
    const float* hs       = (const float*)d_in[0];  // (2,2048,1024)
    const float* in_proj  = (const float*)d_in[1];  // (4096,1024)
    const float* conv_w   = (const float*)d_in[2];  // (2048,4)
    const float* conv_b   = (const float*)d_in[3];  // (2048,)
    const float* x_proj   = (const float*)d_in[4];  // (96,2048)
    const float* dt_proj  = (const float*)d_in[5];  // (2048,64)
    const float* dt_b     = (const float*)d_in[6];  // (2048,)
    const float* A_log    = (const float*)d_in[7];  // (2048,16)
    const float* D_par    = (const float*)d_in[8];  // (2048,)

    float* out  = (float*)d_out;                          // (2,2048,2048)
    float* zout = out + (size_t)BATCH * LEN * DINNER;     // z: (2,2048,2048) as (b,d,l)

    // workspace carve-up (256B aligned)
    char* base = (char*)d_ws;
    size_t off = 0;
    auto alloc = [&](size_t bytes) -> void* {
        void* p = base + off;
        off += (bytes + 255) & ~(size_t)255;
        return p;
    };
    f16*   hs16    = (f16*)  alloc((size_t)NTOK * DMODEL * 2);
    f16*   w1_16   = (f16*)  alloc((size_t)NXZ * DMODEL * 2);
    f16*   xpw16   = (f16*)  alloc((size_t)KDBL * DINNER * 2);
    f16*   dtw16   = (f16*)  alloc((size_t)DINNER * DTRANK * 2);
    float* x_ws    = (float*)alloc((size_t)NTOK * DINNER * 4);   // x (b,d,l)
    float* xc32    = (float*)alloc((size_t)NTOK * DINNER * 4);   // conv out (t,d)
    f16*   xc16    = (f16*)  alloc((size_t)NTOK * DINNER * 2);
    float* zT      = (float*)alloc((size_t)NTOK * DINNER * 4);   // z (t,d)
    float* xdbl32  = (float*)alloc((size_t)NTOK * KDBL * 4);
    f16*   xdbl16  = (f16*)  alloc((size_t)NTOK * DTRANK * 2);
    float* delta_p = (float*)alloc((size_t)NTOK * DINNER * 4);
    (void)ws_size; (void)in_sizes; (void)n_in; (void)out_size;

    const int nHS = NTOK * DMODEL, nW1 = NXZ * DMODEL;
    const int nXP = KDBL * DINNER, nDT = DINNER * DTRANK;
    cvt_f32_f16<<<(nHS + 255) / 256, 256, 0, stream>>>(hs, hs16, nHS);
    cvt_f32_f16<<<(nW1 + 255) / 256, 256, 0, stream>>>(in_proj, w1_16, nW1);
    cvt_f32_f16<<<(nXP + 255) / 256, 256, 0, stream>>>(x_proj, xpw16, nXP);
    cvt_f32_f16<<<(nDT + 255) / 256, 256, 0, stream>>>(dt_proj, dtw16, nDT);

    gemm_inproj<<<1024, 256, 0, stream>>>(hs16, w1_16, x_ws, zout, zT);   // 8192 waves
    conv_silu<<<dim3(64, 64, 2), 256, 0, stream>>>(x_ws, conv_w, conv_b, xc32, xc16);
    gemm_xproj<<<32, 256, 0, stream>>>(xc16, xpw16, xdbl32, xdbl16);      // 256 waves
    gemm_dt<<<512, 256, 0, stream>>>(xdbl16, dtw16, delta_p);             // 4096 waves
    scan_kernel<<<16, 256, 0, stream>>>(delta_p, xc32, xdbl32, zT,
                                        dt_b, A_log, D_par, out);
}
